// G_21406117003961
// MI455X (gfx1250) — matrix operand order repaired and checked by reading, not compiled, not run on hardware
//
#include <hip/hip_runtime.h>
#include <hip/hip_bf16.h>

typedef _Float16 v16h __attribute__((ext_vector_type(16)));
typedef float    v8f  __attribute__((ext_vector_type(8)));

#define NSITES 4096
#define HALFN  2048
#define BATCH  16384
#define SLOPE  0.2f

// ---------------------------------------------------------------------------
// Pack a row-major f32 weight (K x N) into f16 WMMA B-fragment order.
// Block = (ktile, ntile) covering 32(K) x 16(N). Within a block, element
// index = lane*16 + i holds W[ktile*32 + (lane>>4)*16 + i][ntile*16 + (lane&15)]
// so the main kernel's B-frag load is a contiguous 32B per lane.
// ---------------------------------------------------------------------------
__global__ void pack_weights_kernel(const float* __restrict__ src,
                                    _Float16* __restrict__ dst, int K, int N) {
    int tid = blockIdx.x * blockDim.x + threadIdx.x;
    if (tid >= K * N) return;
    int i    = tid & 15;
    int lane = (tid >> 4) & 31;
    int blk  = tid >> 9;
    int ntiles = N >> 4;
    int nt = blk % ntiles;
    int kt = blk / ntiles;
    int k = kt * 32 + ((lane >> 4) << 4) + i;
    int n = nt * 16 + (lane & 15);
    dst[tid] = (_Float16)src[k * N + n];
}

// f[:, IDX_A] = z[:, IDX_A]  (identity passthrough of even-parity sites)
__global__ void copy_a_kernel(const float* __restrict__ z, float* __restrict__ f) {
    int tid = blockIdx.x * blockDim.x + threadIdx.x;   // over BATCH*HALFN
    int r = tid >> 11;
    int k = tid & 2047;
    int lrow = k >> 5, pos = k & 31;
    int site = lrow * 64 + 2 * pos + (lrow & 1);       // parity-0 column
    size_t off = (size_t)r * NSITES + site;
    f[off] = z[off];
}

// ---------------------------------------------------------------------------
// Fused coupling kernel: 4 waves / 128 threads per WG, 64 batch rows per WG,
// 16 rows per wave. All GEMMs via v_wmma_f32_16x16x32_f16.
// Layer-0 z staging is register double-buffered: global loads for chunk kc+1
// are issued before the WMMA burst on chunk kc, so staging latency hides
// behind matrix math instead of a barrier stall.
// ---------------------------------------------------------------------------
__global__ __launch_bounds__(128)
void coupling_kernel(const float* __restrict__ z,
                     const float* __restrict__ s_b2,
                     const float* __restrict__ t_b2,
                     const _Float16* __restrict__ wp,   // packed weights
                     float* __restrict__ f,
                     float* __restrict__ d)
{
    // packed weight layout in d_ws (halves)
    const _Float16* W0p0 = wp + 0;        // s_W0  2048x128
    const _Float16* W0p1 = wp + 262144;   // t_W0
    const _Float16* W1p0 = wp + 524288;   // s_W1  128x128
    const _Float16* W1p1 = wp + 540672;   // t_W1
    const _Float16* W2p0 = wp + 557056;   // s_W2  128x2048
    const _Float16* W2p1 = wp + 819200;   // t_W2

    __shared__ __align__(16) float zbuf[2][64 * 68];   // double-buffered z tiles
    __shared__ _Float16 hbuf[2][64][136];              // activations, f16

    const int tid  = threadIdx.x;
    const int wave = tid >> 5;
    const int lane = tid & 31;
    const int ml   = lane & 15;     // A: M index / B,C: N index
    const int grp  = lane >> 4;     // lane half-group
    const int wgRow = blockIdx.x * 64;
    const int wvRow = wgRow + wave * 16;

    const v8f vzero = {0.f,0.f,0.f,0.f,0.f,0.f,0.f,0.f};

    // per-thread slice of one z chunk: 8 float4 = 32 floats
    float4 stage[8];

    // ---------------- Layer 0: h0 = lrelu(z_a @ W0), K = 2048 ----------------
    v8f acc0[2][8];
    #pragma unroll
    for (int m = 0; m < 2; ++m)
        #pragma unroll
        for (int nt = 0; nt < 8; ++nt) acc0[m][nt] = vzero;

    // prologue: stage chunk 0
    #pragma unroll
    for (int q = 0; q < 8; ++q) {
        int fidx = q * 128 + tid;
        int row = fidx >> 4;
        int c4  = fidx & 15;
        stage[q] = *(const float4*)(z + (size_t)(wgRow + row) * NSITES + c4 * 4);
    }
    #pragma unroll
    for (int q = 0; q < 8; ++q) {
        int fidx = q * 128 + tid;
        *(float4*)&zbuf[0][(fidx >> 4) * 68 + (fidx & 15) * 4] = stage[q];
    }
    __syncthreads();

    for (int kc = 0; kc < 64; ++kc) {          // K-chunk of 32 = one lattice row
        const int p = kc & 1;

        // issue global loads for the NEXT chunk first (latency hidden by WMMAs)
        if (kc + 1 < 64) {
            #pragma unroll
            for (int q = 0; q < 8; ++q) {
                int fidx = q * 128 + tid;
                int row = fidx >> 4;
                int c4  = fidx & 15;
                stage[q] = *(const float4*)(z + (size_t)(wgRow + row) * NSITES
                                            + (kc + 1) * 64 + c4 * 4);
            }
        }

        // A fragment: 16x32 f16, contiguous k order (lane = M + 16*khalf, k = 16*khalf + i)
        v16h a;
        const float* zr = &zbuf[p][(wave * 16 + ml) * 68 + (kc & 1)]; // parity off
        #pragma unroll
        for (int i = 0; i < 16; ++i) {
            int K = i + grp * 16;
            a[i] = (_Float16)zr[2 * K];        // z_a gather: stride-2 in lattice row
        }

        #pragma unroll
        for (int nt = 0; nt < 8; ++nt) {
            v16h b0 = *(const v16h*)(W0p0 + ((kc * 8 + nt) * 512 + lane * 16));
            acc0[0][nt] = __builtin_amdgcn_wmma_f32_16x16x32_f16(
                false, a, false, b0, (short)0, acc0[0][nt], false, false);
            v16h b1 = *(const v16h*)(W0p1 + ((kc * 8 + nt) * 512 + lane * 16));
            acc0[1][nt] = __builtin_amdgcn_wmma_f32_16x16x32_f16(
                false, a, false, b1, (short)0, acc0[1][nt], false, false);
        }

        // commit staged registers into the alternate buffer, then one barrier
        if (kc + 1 < 64) {
            #pragma unroll
            for (int q = 0; q < 8; ++q) {
                int fidx = q * 128 + tid;
                *(float4*)&zbuf[1 - p][(fidx >> 4) * 68 + (fidx & 15) * 4] = stage[q];
            }
        }
        __syncthreads();
    }

    // leaky-relu, spill h0 to LDS (C-layout -> row-major f16)
    #pragma unroll
    for (int m = 0; m < 2; ++m)
        #pragma unroll
        for (int nt = 0; nt < 8; ++nt)
            #pragma unroll
            for (int j = 0; j < 8; ++j) {
                float x = acc0[m][nt][j];
                x = (x > 0.f) ? x : SLOPE * x;
                hbuf[m][wave * 16 + grp * 8 + j][nt * 16 + ml] = (_Float16)x;
            }
    __syncthreads();

    // ---------------- Layer 1: h1 = lrelu(h0 @ W1), K = 128 ----------------
    v8f acc1[2][8];
    #pragma unroll
    for (int m = 0; m < 2; ++m)
        #pragma unroll
        for (int nt = 0; nt < 8; ++nt) acc1[m][nt] = vzero;

    #pragma unroll
    for (int kc = 0; kc < 4; ++kc) {
        #pragma unroll
        for (int m = 0; m < 2; ++m) {
            v16h a;
            #pragma unroll
            for (int i = 0; i < 16; ++i) {
                int K = kc * 32 + i + grp * 16;
                a[i] = hbuf[m][wave * 16 + ml][K];
            }
            #pragma unroll
            for (int nt = 0; nt < 8; ++nt) {
                v16h b = *(const v16h*)((m ? W1p1 : W1p0)
                                        + ((kc * 8 + nt) * 512 + lane * 16));
                acc1[m][nt] = __builtin_amdgcn_wmma_f32_16x16x32_f16(
                    false, a, false, b, (short)0, acc1[m][nt], false, false);
            }
        }
    }
    __syncthreads();

    #pragma unroll
    for (int m = 0; m < 2; ++m)
        #pragma unroll
        for (int nt = 0; nt < 8; ++nt)
            #pragma unroll
            for (int j = 0; j < 8; ++j) {
                float x = acc1[m][nt][j];
                x = (x > 0.f) ? x : SLOPE * x;
                hbuf[m][wave * 16 + grp * 8 + j][nt * 16 + ml] = (_Float16)x;
            }
    __syncthreads();

    // ---------------- Layer 2 + fused elementwise, N = 2048 ----------------
    // Preload both MLPs' A fragments (reused across all 128 N-tiles).
    v16h af[2][4];
    #pragma unroll
    for (int m = 0; m < 2; ++m)
        #pragma unroll
        for (int kc = 0; kc < 4; ++kc)
            #pragma unroll
            for (int i = 0; i < 16; ++i) {
                int K = kc * 32 + i + grp * 16;
                af[m][kc][i] = hbuf[m][wave * 16 + ml][K];
            }

    float dpart[8] = {0.f,0.f,0.f,0.f,0.f,0.f,0.f,0.f};

    for (int nt = 0; nt < 128; ++nt) {
        v8f as = vzero, at = vzero;
        #pragma unroll
        for (int kc = 0; kc < 4; ++kc) {
            v16h bs = *(const v16h*)(W2p0 + ((kc * 128 + nt) * 512 + lane * 16));
            as = __builtin_amdgcn_wmma_f32_16x16x32_f16(
                false, af[0][kc], false, bs, (short)0, as, false, false);
            v16h bt = *(const v16h*)(W2p1 + ((kc * 128 + nt) * 512 + lane * 16));
            at = __builtin_amdgcn_wmma_f32_16x16x32_f16(
                false, af[1][kc], false, bt, (short)0, at, false, false);
        }
        int n = nt * 16 + ml;                   // z_b column index
        float sb = s_b2[n];
        float tb = t_b2[n];
        int lrow = n >> 5, pos = n & 31;
        int siteB = lrow * 64 + 2 * pos + ((lrow & 1) ^ 1);   // parity-1 column
        #pragma unroll
        for (int j = 0; j < 8; ++j) {
            int r = wvRow + grp * 8 + j;        // C-layout row
            float sv = as[j] + sb;              // s_out
            float tv = at[j] + tb;              // t_a
            dpart[j] += sv;
            size_t off = (size_t)r * NSITES + siteB;
            f[off] = (z[off] - tv) * __expf(-sv);
        }
    }

    // d = row-sum of s_out: reduce over the 16-lane N group (wave32-safe)
    #pragma unroll
    for (int j = 0; j < 8; ++j) {
        float v = dpart[j];
        v += __shfl_xor(v, 1, 32);
        v += __shfl_xor(v, 2, 32);
        v += __shfl_xor(v, 4, 32);
        v += __shfl_xor(v, 8, 32);
        if (ml == 0) d[wvRow + grp * 8 + j] = v;
    }
}

extern "C" void kernel_launch(void* const* d_in, const int* in_sizes, int n_in,
                              void* d_out, int out_size, void* d_ws, size_t ws_size,
                              hipStream_t stream) {
    const float* z   = (const float*)d_in[0];
    const float* sW0 = (const float*)d_in[1];
    const float* sW1 = (const float*)d_in[2];
    const float* sW2 = (const float*)d_in[3];
    const float* sb2 = (const float*)d_in[4];
    const float* tW0 = (const float*)d_in[5];
    const float* tW1 = (const float*)d_in[6];
    const float* tW2 = (const float*)d_in[7];
    const float* tb2 = (const float*)d_in[8];

    float* fout = (float*)d_out;
    float* dout = fout + (size_t)BATCH * NSITES;
    _Float16* wp = (_Float16*)d_ws;            // ~2.1 MB of packed f16 weights

    // pack all six weight matrices into WMMA B-fragment order (f16)
    pack_weights_kernel<<<(2048 * 128) / 256, 256, 0, stream>>>(sW0, wp + 0,      2048, 128);
    pack_weights_kernel<<<(2048 * 128) / 256, 256, 0, stream>>>(tW0, wp + 262144, 2048, 128);
    pack_weights_kernel<<<(128 * 128)  / 256, 256, 0, stream>>>(sW1, wp + 524288, 128,  128);
    pack_weights_kernel<<<(128 * 128)  / 256, 256, 0, stream>>>(tW1, wp + 540672, 128,  128);
    pack_weights_kernel<<<(2048 * 128) / 256, 256, 0, stream>>>(sW2, wp + 557056, 128, 2048);
    pack_weights_kernel<<<(2048 * 128) / 256, 256, 0, stream>>>(tW2, wp + 819200, 128, 2048);

    // passthrough of even-parity sites
    copy_a_kernel<<<(BATCH * HALFN) / 256, 256, 0, stream>>>(z, fout);

    // fused MLPs + coupling transform + row-sum
    coupling_kernel<<<BATCH / 64, 128, 0, stream>>>(z, sb2, tb2, wp, fout, dout);
}
